// SelfAttention_32349693674056
// MI455X (gfx1250) — matrix operand order repaired and checked by reading, not compiled, not run on hardware
//
#include <hip/hip_runtime.h>

// ---------------------------------------------------------------------------
// Self-attention forward for MI455X (gfx1250, wave32, WMMA).
//   q = x Wq^T, k = x Wk^T, v = x Wv^T   (bf16 compute, f32 accumulate)
//   att = softmax_causal(q k^T / sqrt(E)) v   (flash-style, online softmax)
//   out = att Wu^T + bu                   (f32 output)
// GEMMs: double-buffered LDS tiles filled by GLOBAL_LOAD_ASYNC_TO_LDS_B128
// (ASYNCcnt path), 8 WMMA per wave per 32-deep K-step.
// ---------------------------------------------------------------------------

typedef __attribute__((ext_vector_type(16))) __bf16 v16bf;
typedef __attribute__((ext_vector_type(8)))  __bf16 v8bf;
typedef __attribute__((ext_vector_type(4)))  __bf16 v4bf;
typedef __attribute__((ext_vector_type(8)))  float  v8f;

#define DEV static __device__ __forceinline__

constexpr int   Bc = 4, Sc = 2048, Ec = 1280, Hc = 5, Dc = 256;
constexpr int   Mc = Bc * Sc;                       // 8192 flattened rows
constexpr float SCALE = 0.027950849718747374f;      // 1/sqrt(1280)

DEV int lane_id() { return (int)(threadIdx.x & 31u); }

DEV v8f wmma_bf16(v16bf a, v16bf b, v8f c) {
  // D = A(16x32) * B(32x16) + C, f32 accumulate
  return __builtin_amdgcn_wmma_f32_16x16x32_bf16(false, a, false, b,
                                                 (short)0, c, false, false);
}

// A fragment (16 rows x 32 K, bf16) from a row-major source, leading dim `ld`
// Layout: lanes 0-15 -> row lane, K 0..15;
//         lanes 16-31 -> row lane-16, K 16..31
DEV v16bf frag_a(const __bf16* p, int ld) {
  const int l = lane_id();
  const __bf16* r = p + (l & 15) * ld + ((l >> 4) << 4);
  v8bf lo = *(const v8bf*)(r);        // K 0..7  (or 16..23)
  v8bf hi = *(const v8bf*)(r + 8);    // K 8..15 (or 24..31)
  return __builtin_shufflevector(lo, hi, 0, 1, 2, 3, 4, 5, 6, 7,
                                         8, 9, 10, 11, 12, 13, 14, 15);
}

// B fragment (32 K x 16 cols, bf16). Source is B^T row-major (row = out col,
// contiguous along K): lanes 0-15 -> col lane, K 0..15; lanes 16-31 -> col
// lane-16, K 16..31.  One 32-byte contiguous load per lane.
DEV v16bf frag_b(const __bf16* p, int ld) {
  const int l = lane_id();
  return *(const v16bf*)(p + (l & 15) * ld + ((l >> 4) << 4));
}

// xor-shuffle within 16-lane halves (ds_swizzle group-of-32 encoding)
template <int XM> DEV float swz_xor(float x) {
  return __int_as_float(
      __builtin_amdgcn_ds_swizzle(__float_as_int(x), (XM << 10) | 0x1f));
}
DEV float rowmax16(float x) {
  x = fmaxf(x, swz_xor<1>(x));
  x = fmaxf(x, swz_xor<2>(x));
  x = fmaxf(x, swz_xor<4>(x));
  x = fmaxf(x, swz_xor<8>(x));
  return x;
}
DEV float rowsum16(float x) {
  x += swz_xor<1>(x);
  x += swz_xor<2>(x);
  x += swz_xor<4>(x);
  x += swz_xor<8>(x);
  return x;
}

// ---------------------------------------------------------------------------
// fp32 -> bf16 conversion (4 elements / thread, n % 1024 == 0 for all inputs)
// ---------------------------------------------------------------------------
__global__ __launch_bounds__(256) void k_cvt(const float* __restrict__ in,
                                             __bf16* __restrict__ out, int n) {
  int i = (int)(blockIdx.x * 256 + threadIdx.x) * 4;
  if (i < n) {
    float4 f = *(const float4*)(in + i);
    v4bf o;
    o[0] = (__bf16)f.x; o[1] = (__bf16)f.y;
    o[2] = (__bf16)f.z; o[3] = (__bf16)f.w;
    *(v4bf*)(out + i) = o;
  }
}

// ---------------------------------------------------------------------------
// Shared GEMM main loop: acc(128x128 block) += X[m0:,k] @ W[n0:,k]^T.
// Double-buffered 128x32 LDS tiles (stride 48 elems = 96 B, 32 B-aligned
// rows). Tiles are filled by async direct-to-LDS loads: per wave 4 x
// global_load_async_to_lds_b128 (each moves 8 rows x 64 B), waited with
// s_wait_asynccnt before the single per-step barrier.
// ---------------------------------------------------------------------------
DEV void gemm_core(const __bf16* __restrict__ X, const __bf16* __restrict__ W,
                   int m0, int n0, __bf16* As, __bf16* Bs, v8f (&acc)[4][2]) {
  const int tid  = (int)threadIdx.x;
  const int lane = tid & 31;
  const int wave = tid >> 5;
  const int wm = wave >> 2;          // 0..1 (M half)
  const int wn = wave & 3;           // 0..3 (N quarter)

  // async staging geometry: lane -> (row, 16B chunk) of this wave's 16 rows
  const int rr = lane >> 2;          // 0..7
  const int ce = (lane & 3) << 3;    // element col 0,8,16,24
  const int row = wave * 16 + rr;    // 0..127 (+8 for second op)

  const __bf16* gA = X + (m0 + row) * Ec + ce;
  const __bf16* gB = W + (n0 + row) * Ec + ce;
  const unsigned lA = (unsigned)(size_t)(As + row * 48 + ce);
  const unsigned lB = (unsigned)(size_t)(Bs + row * 48 + ce);
  constexpr unsigned BUFB = 128u * 48u * 2u;   // bytes per LDS buffer
  constexpr int      BUFE = 128 * 48;          // elements per LDS buffer

  auto stage = [&](int buf, int k0) {
    const __bf16* ga = gA + k0;
    const __bf16* gb = gB + k0;
    const unsigned la = lA + (unsigned)buf * BUFB;
    const unsigned lb = lB + (unsigned)buf * BUFB;
    asm volatile("global_load_async_to_lds_b128 %0, %1, off"
                 :: "v"(la), "v"(ga) : "memory");
    asm volatile("global_load_async_to_lds_b128 %0, %1, off"
                 :: "v"(la + 8u * 96u), "v"(ga + 8 * Ec) : "memory");
    asm volatile("global_load_async_to_lds_b128 %0, %1, off"
                 :: "v"(lb), "v"(gb) : "memory");
    asm volatile("global_load_async_to_lds_b128 %0, %1, off"
                 :: "v"(lb + 8u * 96u), "v"(gb + 8 * Ec) : "memory");
  };

  stage(0, 0);
  asm volatile("s_wait_asynccnt 0x0" ::: "memory");
  __syncthreads();

  int buf = 0;
  for (int k0 = 0; k0 < Ec; k0 += 32) {
    if (k0 + 32 < Ec) stage(buf ^ 1, k0 + 32);   // prefetch next tile -> LDS

    const __bf16* Ab = As + buf * BUFE;
    const __bf16* Bb = Bs + buf * BUFE;
    v16bf b0 = frag_b(Bb + (wn * 32) * 48, 48);
    v16bf b1 = frag_b(Bb + (wn * 32 + 16) * 48, 48);
#pragma unroll
    for (int i = 0; i < 4; ++i) {
      v16bf a = frag_a(Ab + (wm * 64 + i * 16) * 48, 48);
      acc[i][0] = wmma_bf16(a, b0, acc[i][0]);
      acc[i][1] = wmma_bf16(a, b1, acc[i][1]);
    }

    asm volatile("s_wait_asynccnt 0x0" ::: "memory");
    __syncthreads();
    buf ^= 1;
  }
}

// ---------------------------------------------------------------------------
// GEMM O = X @ W^T with bf16 output scattered head-major:
//   vtrans=0 -> [B,H,S,D] (Q, K);  vtrans=1 -> [B,H,D,S] (V transposed)
// ---------------------------------------------------------------------------
__global__ __launch_bounds__(256)
void k_gemm_qkv(const __bf16* __restrict__ X, const __bf16* __restrict__ W,
                __bf16* __restrict__ out, int vtrans) {
  __shared__ __attribute__((aligned(32))) __bf16 As[2][128 * 48];
  __shared__ __attribute__((aligned(32))) __bf16 Bs[2][128 * 48];

  const int m0 = blockIdx.y * 128;
  const int n0 = blockIdx.x * 128;
  const int lane = (int)threadIdx.x & 31;
  const int wm = ((int)threadIdx.x >> 5) >> 2;
  const int wn = ((int)threadIdx.x >> 5) & 3;

  v8f acc[4][2] = {};
  gemm_core(X, W, m0, n0, &As[0][0], &Bs[0][0], acc);

#pragma unroll
  for (int i = 0; i < 4; ++i) {
#pragma unroll
    for (int j = 0; j < 2; ++j) {
      const int col  = n0 + wn * 32 + j * 16 + (lane & 15);
      const int rowb = m0 + wm * 64 + i * 16 + ((lane >> 4) << 3);
      const int h = col >> 8, dd = col & 255;
#pragma unroll
      for (int v = 0; v < 8; ++v) {
        const int m = rowb + v;
        const int b = m >> 11, s = m & 2047;
        const __bf16 val = (__bf16)acc[i][j][v];
        if (vtrans)
          out[((b * Hc + h) * Dc + dd) * Sc + s] = val;   // V^T: [B,H,D,S]
        else
          out[((b * Hc + h) * Sc + s) * Dc + dd] = val;   // Q/K: [B,H,S,D]
      }
    }
  }
}

// ---------------------------------------------------------------------------
// Final projection: out = attOut @ Wu^T + bu, f32 output [Mc x Ec].
// ---------------------------------------------------------------------------
__global__ __launch_bounds__(256)
void k_gemm_out(const __bf16* __restrict__ X, const __bf16* __restrict__ W,
                const float* __restrict__ bias, float* __restrict__ out) {
  __shared__ __attribute__((aligned(32))) __bf16 As[2][128 * 48];
  __shared__ __attribute__((aligned(32))) __bf16 Bs[2][128 * 48];

  const int m0 = blockIdx.y * 128;
  const int n0 = blockIdx.x * 128;
  const int lane = (int)threadIdx.x & 31;
  const int wm = ((int)threadIdx.x >> 5) >> 2;
  const int wn = ((int)threadIdx.x >> 5) & 3;

  v8f acc[4][2] = {};
  gemm_core(X, W, m0, n0, &As[0][0], &Bs[0][0], acc);

#pragma unroll
  for (int i = 0; i < 4; ++i) {
#pragma unroll
    for (int j = 0; j < 2; ++j) {
      const int col  = n0 + wn * 32 + j * 16 + (lane & 15);
      const int rowb = m0 + wm * 64 + i * 16 + ((lane >> 4) << 3);
      const float bv = bias[col];
#pragma unroll
      for (int v = 0; v < 8; ++v)
        out[(rowb + v) * Ec + col] = acc[i][j][v] + bv;
    }
  }
}

// ---------------------------------------------------------------------------
// Flash attention, causal. One wave owns 16 query rows; loops over 32-key
// tiles with online softmax. Scores: A=Q (global, WGP$-resident 8KB/wave),
// B=K rows. P is bounced through per-wave LDS scratch to become an A operand
// for P @ V, with V pre-transposed so B fragments are contiguous.
// O accumulator: 16 x (16x16 f32) tiles resident in VGPRs. K/V stay in the
// 192 MB L2 (2 MB per head); next K tile is prefetched each iteration.
// ---------------------------------------------------------------------------
__global__ __launch_bounds__(256)
void k_attn(const __bf16* __restrict__ Q,   // [B,H,S,D]
            const __bf16* __restrict__ K,   // [B,H,S,D]
            const __bf16* __restrict__ Vt,  // [B,H,D,S]
            __bf16* __restrict__ O) {       // [B,S,E] bf16
  __shared__ __attribute__((aligned(32))) __bf16 Ps[8 * 16 * 48];

  const int lane  = (int)threadIdx.x & 31;
  const int wave  = (int)threadIdx.x >> 5;
  const int gid   = blockIdx.x * 8 + wave;   // 0 .. B*H*(S/16)-1
  const int qtile = gid & 127;               // S/16 = 128
  const int bh    = gid >> 7;
  const int qbase = qtile << 4;

  const __bf16* qp = Q + (size_t)bh * Sc * Dc + qbase * Dc;
  const __bf16* kp = K + (size_t)bh * Sc * Dc;
  const __bf16* vp = Vt + (size_t)bh * Dc * Sc;
  __bf16* ps = Ps + wave * (16 * 48);

  v8f o[16] = {};
  v8f mrun, lrun;
#pragma unroll
  for (int v = 0; v < 8; ++v) { mrun[v] = -1e30f; lrun[v] = 0.0f; }

  const int mrow0 = (lane >> 4) << 3;   // row offset of this half-wave in C

  for (int jb = 0; jb <= qbase + 15; jb += 32) {
    // prefetch next 32 K rows toward WGP while this tile computes
    if (jb + 32 <= qbase + 15) {
      __builtin_prefetch(kp + (size_t)(jb + 32 + lane) * Dc, 0, 1);
      __builtin_prefetch(kp + (size_t)(jb + 32 + lane) * Dc + 128, 0, 1);
    }

    // ---- scores S = Q K^T for 16 q-rows x 32 keys (two 16x16 C tiles) ----
    v8f s0 = {}, s1 = {};
#pragma unroll
    for (int dk = 0; dk < 8; ++dk) {           // D = 256 = 8 * 32
      v16bf a  = frag_a(qp + dk * 32, Dc);
      v16bf b0 = frag_b(kp + (jb) * Dc + dk * 32, Dc);
      v16bf b1 = frag_b(kp + (jb + 16) * Dc + dk * 32, Dc);
      s0 = wmma_bf16(a, b0, s0);
      s1 = wmma_bf16(a, b1, s1);
    }

    // ---- scale, causal mask, online softmax update ----
    v8f sc;
#pragma unroll
    for (int v = 0; v < 8; ++v) {
      float x0 = s0[v] * SCALE;
      float x1 = s1[v] * SCALE;
      const int qrow = qbase + mrow0 + v;
      const int c0 = jb + (lane & 15);
      if (c0 > qrow)      x0 = -1e30f;     // strictly-upper -> -inf
      if (c0 + 16 > qrow) x1 = -1e30f;
      float mt   = rowmax16(fmaxf(x0, x1));
      float mnew = fmaxf(mrun[v], mt);
      sc[v] = __expf(mrun[v] - mnew);
      float p0 = __expf(x0 - mnew);
      float p1 = __expf(x1 - mnew);
      lrun[v] = lrun[v] * sc[v] + rowsum16(p0 + p1);
      mrun[v] = mnew;
      // C-layout -> row-major LDS scratch (bf16) for re-fragmentation as A
      const int mr = mrow0 + v;
      ps[mr * 48 + (lane & 15)]      = (__bf16)p0;
      ps[mr * 48 + (lane & 15) + 16] = (__bf16)p1;
    }

    // ---- rescale accumulator by exp(m_old - m_new), per row ----
#pragma unroll
    for (int t = 0; t < 16; ++t)
#pragma unroll
      for (int v = 0; v < 8; ++v) o[t][v] *= sc[v];

    // ---- O += P @ V  (P: 16x32, V: 32x256 via Vt rows) ----
    v16bf pa = frag_a(ps, 48);
#pragma unroll
    for (int t = 0; t < 16; ++t) {
      v16bf bv = frag_b(vp + (t * 16) * Sc + jb, Sc);
      o[t] = wmma_bf16(pa, bv, o[t]);
    }
  }

  // ---- normalize and store to [B, S, E] (E-slot = h*256 + dd) ----
  const int b = bh / Hc, h = bh % Hc;
#pragma unroll
  for (int v = 0; v < 8; ++v) lrun[v] = 1.0f / lrun[v];
  __bf16* op = O + ((size_t)b * Sc + qbase) * Ec + h * Dc;
#pragma unroll
  for (int t = 0; t < 16; ++t) {
    const int col = t * 16 + (lane & 15);
#pragma unroll
    for (int v = 0; v < 8; ++v)
      op[(mrow0 + v) * Ec + col] = (__bf16)(o[t][v] * lrun[v]);
  }
}

// ---------------------------------------------------------------------------
// Host launcher
// ---------------------------------------------------------------------------
extern "C" void kernel_launch(void* const* d_in, const int* in_sizes, int n_in,
                              void* d_out, int out_size, void* d_ws,
                              size_t ws_size, hipStream_t stream) {
  (void)in_sizes; (void)n_in; (void)out_size; (void)ws_size;

  const float* x  = (const float*)d_in[0];
  const float* Wq = (const float*)d_in[1];
  const float* Wk = (const float*)d_in[2];
  const float* Wv = (const float*)d_in[3];
  const float* Wu = (const float*)d_in[4];
  const float* bu = (const float*)d_in[5];
  float* out = (float*)d_out;

  // workspace layout (bf16), all 32 B aligned: ~118 MB total
  __bf16* xb  = (__bf16*)d_ws;                 // [Mc, Ec]
  __bf16* wqb = xb  + (size_t)Mc * Ec;         // [Ec, Ec]
  __bf16* wkb = wqb + (size_t)Ec * Ec;
  __bf16* wvb = wkb + (size_t)Ec * Ec;
  __bf16* wub = wvb + (size_t)Ec * Ec;
  __bf16* q   = wub + (size_t)Ec * Ec;         // [B,H,S,D]
  __bf16* k   = q   + (size_t)Mc * Ec;         // [B,H,S,D]
  __bf16* vt  = k   + (size_t)Mc * Ec;         // [B,H,D,S]
  __bf16* ao  = vt  + (size_t)Mc * Ec;         // [B,S,E] attention output

  const int nx = Mc * Ec;   // 10485760
  const int nw = Ec * Ec;   // 1638400
  k_cvt<<<nx / 1024, 256, 0, stream>>>(x, xb, nx);
  k_cvt<<<nw / 1024, 256, 0, stream>>>(Wq, wqb, nw);
  k_cvt<<<nw / 1024, 256, 0, stream>>>(Wk, wkb, nw);
  k_cvt<<<nw / 1024, 256, 0, stream>>>(Wv, wvb, nw);
  k_cvt<<<nw / 1024, 256, 0, stream>>>(Wu, wub, nw);

  dim3 g(Ec / 128, Mc / 128);   // (10, 64)
  k_gemm_qkv<<<g, 256, 0, stream>>>(xb, wqb, q, 0);
  k_gemm_qkv<<<g, 256, 0, stream>>>(xb, wkb, k, 0);
  k_gemm_qkv<<<g, 256, 0, stream>>>(xb, wvb, vt, 1);

  k_attn<<<(Bc * Hc * (Sc / 16)) / 8, 256, 0, stream>>>(q, k, vt, ao);

  k_gemm_out<<<g, 256, 0, stream>>>(ao, wub, bu, out);
}
